// ODEOneSeqDecoder_2345052143986
// MI455X (gfx1250) — matrix operand order repaired and checked by reading, not compiled, not run on hardware
//
#include <hip/hip_runtime.h>

typedef _Float16 h16 __attribute__((ext_vector_type(16)));
typedef _Float16 h8  __attribute__((ext_vector_type(8)));
typedef _Float16 h4  __attribute__((ext_vector_type(4)));
typedef float    f8  __attribute__((ext_vector_type(8)));

#define TSTEPS 512
#define BATCH  1024
#define COSK   0.47746483f  /* 3/(2*pi): cos(3x) = v_cos(x*COSK) */

// Cross-lane data + W1a fragments: 52,224 bytes LDS. One 16-row tile per block,
// two waves cooperating on it (N-dimension split).
struct __align__(16) Smem {
  _Float16 frag1[8][32][16];  // W1a B fragments (per-lane rows), kc*2+nt
  float    c0[32][16];        // layer-1 C operand, TRANSPOSED: c0[n][m]
  float    y [16][128];       // state (fp32)
  float    k1[16][128];
  float    k2[16][128];
  float    k3[16][128];       // doubles as latent tile during init
  _Float16 ev [16][128];      // f16 eval point (A source, layer 1)
  _Float16 evP[16][64];       // f16 snapshot of current y pos (A source, traj head)
  _Float16 h1m[16][32];       // layer-1 activations (A source, layer 2)
  _Float16 h2m[16][64];       // layer-2 activations (A source, layer 3)
};

static __device__ __forceinline__ h16 cat16(h8 lo, h8 hi) {
  h16 r;
#pragma unroll
  for (int i = 0; i < 8; ++i) { r[i] = lo[i]; r[8 + i] = hi[i]; }
  return r;
}

// A-matrix (16-bit, 16xK): row = lane&15, kbase = (lane>>4)*16;
// halves 0..15 -> K = kOff+kbase+0..15.
static __device__ __forceinline__ h16 loadA(const _Float16* base, int rs, int kOff, int lane) {
  const int row = lane & 15, kb = (lane >> 4) << 4;
  const _Float16* p = base + row * rs + kOff + kb;
  return cat16(*(const h8*)p, *(const h8*)(p + 8));
}

static __device__ __forceinline__ h16 loadBfrag(const _Float16* p) {
  return cat16(((const h8*)p)[0], ((const h8*)p)[1]);
}

static __device__ __forceinline__ f8 wmma16(h16 a, h16 b, f8 c) {
  return __builtin_amdgcn_wmma_f32_16x16x32_f16(false, a, false, b, (short)0, c, false, false);
}

// MLP, N-split across 2 waves:
//  wave w: L1 N-tile w; L2 N-tiles {2w,2w+1}; L3 N-tiles {2w,2w+1}.
// kdst!=null: kdst[:,64+n] = acc.  kdst==null: y[:,64+n] += dt8*acc.
static __device__ __forceinline__ void runNet(
    Smem& s, int lane, int wid,
    const h16 (&fW2loc)[2], const h16 (&fW3loc)[4],
    f8 c0loc, const float (&b2loc)[2], const float (&b3loc)[2],
    float (*kdst)[128], float dt8) {
  const int nn = lane & 15, mh = (lane >> 4) << 3;
  // ---- layer 1: (16x128) @ (128x32), C = c0; this wave's N-tile = wid ----
  {
    f8 d0 = c0loc;
#pragma unroll
    for (int kc = 0; kc < 4; ++kc) {
      h16 a = loadA(&s.ev[0][0], 128, kc * 32, lane);
      d0 = wmma16(a, loadBfrag(&s.frag1[kc * 2 + wid][lane][0]), d0);
    }
#pragma unroll
    for (int i = 0; i < 8; ++i)
      s.h1m[mh + i][wid * 16 + nn] = (_Float16)__builtin_amdgcn_cosf(COSK * d0[i]);
  }
  __syncthreads();
  // ---- layer 2: (16x32) @ (32x64), C = b2; N-tiles 2w, 2w+1 ----
  {
    h16 a = loadA(&s.h1m[0][0], 32, 0, lane);
    f8 e0, e1;
#pragma unroll
    for (int i = 0; i < 8; ++i) { e0[i] = b2loc[0]; e1[i] = b2loc[1]; }
    e0 = wmma16(a, fW2loc[0], e0);
    e1 = wmma16(a, fW2loc[1], e1);
    const int n0 = wid * 32 + nn;
#pragma unroll
    for (int i = 0; i < 8; ++i) {
      s.h2m[mh + i][n0]      = (_Float16)__builtin_amdgcn_cosf(COSK * e0[i]);
      s.h2m[mh + i][n0 + 16] = (_Float16)__builtin_amdgcn_cosf(COSK * e1[i]);
    }
  }
  __syncthreads();
  // ---- layer 3: (16x64) @ (64x64), C = b3; N-tiles 2w, 2w+1 ----
  {
    f8 g0, g1;
#pragma unroll
    for (int i = 0; i < 8; ++i) { g0[i] = b3loc[0]; g1[i] = b3loc[1]; }
#pragma unroll
    for (int kc = 0; kc < 2; ++kc) {
      h16 a = loadA(&s.h2m[0][0], 64, kc * 32, lane);
      g0 = wmma16(a, fW3loc[kc * 2 + 0], g0);
      g1 = wmma16(a, fW3loc[kc * 2 + 1], g1);
    }
    const int n0 = 64 + wid * 32 + nn;
    if (kdst) {
#pragma unroll
      for (int i = 0; i < 8; ++i) {
        kdst[mh + i][n0]      = g0[i];
        kdst[mh + i][n0 + 16] = g1[i];
      }
    } else {
#pragma unroll
      for (int i = 0; i < 8; ++i) {
        s.y[mh + i][n0]      += dt8 * g0[i];
        s.y[mh + i][n0 + 16] += dt8 * g1[i];
      }
    }
  }
}

// e = y + c1*k1 + c2*k2 -> ev (f16), float4-vectorized across 64 lanes:
// lane owns column group tid&31 (4 floats); rows split by wave (8 each).
static __device__ __forceinline__ void buildEval(Smem& s, int tid, int nk,
                                                 float c1, float c2, float (*kv)[128]) {
  const int cg = tid & 31, wid = tid >> 5;
  const float4* Y  = (const float4*)&s.y[0][0];
  const float4* K1 = (const float4*)&s.k1[0][0];
  const float4* K2 = (const float4*)&s.k2[0][0];
  float4* KV = (float4*)&kv[0][0];
#pragma unroll 2
  for (int it = 0; it < 8; ++it) {
    const int m = (wid << 3) + it;
    const int v = m * 32 + cg;
    float4 e = Y[v];
    if (nk >= 1) { float4 a = K1[v]; e.x += c1 * a.x; e.y += c1 * a.y; e.z += c1 * a.z; e.w += c1 * a.w; }
    if (nk >= 2) { float4 a = K2[v]; e.x += c2 * a.x; e.y += c2 * a.y; e.z += c2 * a.z; e.w += c2 * a.w; }
    h4 hv; hv[0] = (_Float16)e.x; hv[1] = (_Float16)e.y; hv[2] = (_Float16)e.z; hv[3] = (_Float16)e.w;
    *(h4*)&s.ev[m][cg << 2] = hv;
    if (cg >= 16) KV[m * 32 + (cg - 16)] = e;  // k[:, :64] = vel of eval point
  }
}

// e4 = y + dt*(k1 - k2 + k3); k4_pos = e4_vel; finalize pos cols of y_new
// (+ f16 snapshot into evP); vel cols partially updated (acc4 added by
// runNet(nullptr, dt8)). Vel-first to cut pressure; 4 iterations per lane.
static __device__ __forceinline__ void pass4(Smem& s, int tid, float dt, float dt8) {
  float4* Y  = (float4*)&s.y[0][0];
  const float4* K1 = (const float4*)&s.k1[0][0];
  const float4* K2 = (const float4*)&s.k2[0][0];
  const float4* K3 = (const float4*)&s.k3[0][0];
  const int j = tid & 15, r = tid >> 4;  // r in 0..3
#pragma unroll 1
  for (int it = 0; it < 4; ++it) {
    const int m = r + (it << 2);
    const int ip = m * 32 + j, iv = ip + 16;
    // --- vel half ---
    float4 yv = Y[iv], av = K1[iv], bv = K2[iv], cv = K3[iv];
    float4 e4v, nv;
    e4v.x = yv.x + dt * (av.x - bv.x + cv.x);  nv.x = yv.x + dt8 * (av.x + 3.0f * (bv.x + cv.x));
    e4v.y = yv.y + dt * (av.y - bv.y + cv.y);  nv.y = yv.y + dt8 * (av.y + 3.0f * (bv.y + cv.y));
    e4v.z = yv.z + dt * (av.z - bv.z + cv.z);  nv.z = yv.z + dt8 * (av.z + 3.0f * (bv.z + cv.z));
    e4v.w = yv.w + dt * (av.w - bv.w + cv.w);  nv.w = yv.w + dt8 * (av.w + 3.0f * (bv.w + cv.w));
    h4 hv; hv[0] = (_Float16)e4v.x; hv[1] = (_Float16)e4v.y; hv[2] = (_Float16)e4v.z; hv[3] = (_Float16)e4v.w;
    *(h4*)&s.ev[m][64 + (j << 2)] = hv;
    Y[iv] = nv;
    // --- pos half (uses e4v as k4_pos) ---
    float4 yp = Y[ip], ap = K1[ip], bp = K2[ip], cp = K3[ip];
    float4 e4p, np;
    e4p.x = yp.x + dt * (ap.x - bp.x + cp.x);  np.x = yp.x + dt8 * (ap.x + 3.0f * (bp.x + cp.x) + e4v.x);
    e4p.y = yp.y + dt * (ap.y - bp.y + cp.y);  np.y = yp.y + dt8 * (ap.y + 3.0f * (bp.y + cp.y) + e4v.y);
    e4p.z = yp.z + dt * (ap.z - bp.z + cp.z);  np.z = yp.z + dt8 * (ap.z + 3.0f * (bp.z + cp.z) + e4v.z);
    e4p.w = yp.w + dt * (ap.w - bp.w + cp.w);  np.w = yp.w + dt8 * (ap.w + 3.0f * (bp.w + cp.w) + e4v.w);
    h4 hp;  hp[0]  = (_Float16)e4p.x; hp[1]  = (_Float16)e4p.y; hp[2]  = (_Float16)e4p.z; hp[3]  = (_Float16)e4p.w;
    h4 hnp; hnp[0] = (_Float16)np.x;  hnp[1] = (_Float16)np.y;  hnp[2] = (_Float16)np.z;  hnp[3] = (_Float16)np.w;
    *(h4*)&s.ev[m][j << 2]  = hp;
    *(h4*)&s.evP[m][j << 2] = hnp;  // final new pos -> traj head A source
    Y[ip] = np;
  }
}

// traj head (wave 0 only): evP(16x64) @ Wobs(64x16 zero-padded) + bobs, 2 WMMAs.
// N-column 0/1 lanes store traj_x, column 2 stores sigmoid, column 3 zeros logpt.
static __device__ __forceinline__ void writeOutTraj(Smem& s, int lane, int row0, int tIdx,
                                                    float* out, const h16 (&fWo)[2], float bobsr) {
  const int nn = lane & 15, mh = (lane >> 4) << 3;
  f8 d;
#pragma unroll
  for (int i = 0; i < 8; ++i) d[i] = bobsr;
#pragma unroll
  for (int kc = 0; kc < 2; ++kc)
    d = wmma16(loadA(&s.evP[0][0], 64, kc * 32, lane), fWo[kc], d);
  const size_t TX = (size_t)TSTEPS * BATCH * 2;
  const size_t TP = TX + (size_t)TSTEPS * BATCH;
  const size_t base = (size_t)tIdx * BATCH + (size_t)row0 + (size_t)mh;
  if (nn == 0) {
#pragma unroll
    for (int i = 0; i < 8; ++i) out[(base + i) * 2 + 0] = d[i];
  } else if (nn == 1) {
#pragma unroll
    for (int i = 0; i < 8; ++i) out[(base + i) * 2 + 1] = d[i];
  } else if (nn == 2) {
    // sigmoid via hardware fast reciprocal (v_rcp_f32), not IEEE divide
#pragma unroll
    for (int i = 0; i < 8; ++i) out[TX + base + i] = __builtin_amdgcn_rcpf(1.0f + __expf(-d[i]));
  } else if (nn == 3) {
#pragma unroll
    for (int i = 0; i < 8; ++i) out[TP + base + i] = 0.0f;  // logpt
  }
}

__global__ void __launch_bounds__(64) ode_rollout_kernel(
    const float* __restrict__ latent, const float* __restrict__ tArr,
    const float* __restrict__ W_l2d, const float* __restrict__ b_l2d,
    const float* __restrict__ W1, const float* __restrict__ b1,
    const float* __restrict__ W2, const float* __restrict__ b2,
    const float* __restrict__ W3, const float* __restrict__ b3,
    const float* __restrict__ W_obs, const float* __restrict__ b_obs,
    float* __restrict__ out) {
  __shared__ Smem s;
  const int tid = threadIdx.x;           // 0..63
  const int lane = tid & 31, wid = tid >> 5;
  const int row0 = blockIdx.x * 16;
  const int nn = lane & 15, mh = (lane >> 4) << 3;
  const int kh = (lane >> 4) << 4;

  // ---- B fragments (per-lane data). Each wave keeps only ITS N-tiles:
  // wave w: L2 tiles {2w,2w+1}, L3 tiles {2w,2w+1}. W1a fragments -> LDS.
  h16 fW2loc[2], fW3loc[4], fWo[2];
  {
    if (wid == 0) {
#pragma unroll
      for (int f = 0; f < 8; ++f) {  // W1a -> LDS (first 128 rows of W1: pos|vel)
        const int kc = f >> 1, nt = f & 1, n = nn + nt * 16, k0 = kc * 32 + kh;
#pragma unroll
        for (int i = 0; i < 16; ++i) s.frag1[f][lane][i] = (_Float16)W1[(k0 + i) * 32 + n];
      }
    }
#pragma unroll
    for (int j = 0; j < 2; ++j) {  // W2, this wave's N-tiles
      const int n = nn + (wid * 2 + j) * 16;
#pragma unroll
      for (int i = 0; i < 16; ++i) fW2loc[j][i] = (_Float16)W2[(kh + i) * 64 + n];
    }
#pragma unroll
    for (int f = 0; f < 4; ++f) {  // W3, this wave's N-tiles
      const int kc = f >> 1, j = f & 1, n = nn + (wid * 2 + j) * 16, k0 = kc * 32 + kh;
#pragma unroll
      for (int i = 0; i < 16; ++i) fW3loc[f][i] = (_Float16)W3[(k0 + i) * 64 + n];
    }
#pragma unroll
    for (int kc = 0; kc < 2; ++kc) {  // W_obs zero-padded to N=16 (wave 0 uses)
#pragma unroll
      for (int i = 0; i < 16; ++i)
        fWo[kc][i] = (nn < 3) ? (_Float16)W_obs[(kc * 32 + kh + i) * 3 + nn] : (_Float16)0.0f;
    }
  }
  // Bias splats for this wave's layer-2/3 tiles + traj head.
  float b2loc[2], b3loc[2];
#pragma unroll
  for (int j = 0; j < 2; ++j) {
    b2loc[j] = b2[(wid * 2 + j) * 16 + nn];
    b3loc[j] = b3[(wid * 2 + j) * 16 + nn];
  }
  const float bobsr = (nn < 3) ? b_obs[nn] : 0.0f;

  // latent tile parked in k3 during init
  for (int idx = tid; idx < 2048; idx += 64) {
    const int m = idx >> 7, c = idx & 127;
    s.k3[m][c] = latent[(size_t)(row0 + m) * 128 + c];
  }
  __syncthreads();

  // y0 = latent @ W_l2d + b_l2d   (once, scalar, 64 lanes)
  for (int idx = tid; idx < 2048; idx += 64) {
    const int m = idx >> 7, n = idx & 127;
    float acc = b_l2d[n];
    for (int k = 0; k < 128; ++k) acc += s.k3[m][k] * W_l2d[k * 128 + n];
    s.y[m][n] = acc;
  }
  // c0 = latent @ W1[128:256,:] + b1, stored TRANSPOSED: c0[n][m]
  for (int idx = tid; idx < 512; idx += 64) {
    const int m = idx & 15, n = idx >> 4;
    float acc = b1[n];
    for (int k = 0; k < 128; ++k) acc += s.k3[m][k] * W1[(128 + k) * 32 + n];
    s.c0[n][m] = acc;
  }
  // f16 snapshot of y0 pos for the traj head
  for (int idx = tid; idx < 1024; idx += 64) {
    const int m = idx >> 6, c = idx & 63;
    s.evP[m][c] = (_Float16)s.y[m][c];
  }
  __syncthreads();

  // Layer-1 C fragment: contiguous column of transposed c0 -> 2x ds_load_b128.
  f8 c0loc;
  {
    const float4 lo = *(const float4*)&s.c0[wid * 16 + nn][mh];
    const float4 hi = *(const float4*)&s.c0[wid * 16 + nn][mh + 4];
    c0loc[0] = lo.x; c0loc[1] = lo.y; c0loc[2] = lo.z; c0loc[3] = lo.w;
    c0loc[4] = hi.x; c0loc[5] = hi.y; c0loc[6] = hi.z; c0loc[7] = hi.w;
  }

  if (wid == 0) writeOutTraj(s, lane, row0, 0, out, fWo, bobsr);  // seq[0] = y0
  __syncthreads();

  for (int st = 0; st < TSTEPS - 1; ++st) {
    const float dt = tArr[st + 1] - tArr[st];
    const float dt3 = dt * (1.0f / 3.0f);
    const float dt8 = dt * 0.125f;

    buildEval(s, tid, 0, 0.0f, 0.0f, s.k1);
    __syncthreads();
    runNet(s, lane, wid, fW2loc, fW3loc, c0loc, b2loc, b3loc, s.k1, 0.0f);
    __syncthreads();
    buildEval(s, tid, 1, dt3, 0.0f, s.k2);
    __syncthreads();
    runNet(s, lane, wid, fW2loc, fW3loc, c0loc, b2loc, b3loc, s.k2, 0.0f);
    __syncthreads();
    buildEval(s, tid, 2, -dt3, dt, s.k3);
    __syncthreads();
    runNet(s, lane, wid, fW2loc, fW3loc, c0loc, b2loc, b3loc, s.k3, 0.0f);
    __syncthreads();
    pass4(s, tid, dt, dt8);
    __syncthreads();
    runNet(s, lane, wid, fW2loc, fW3loc, c0loc, b2loc, b3loc, nullptr, dt8);
    __syncthreads();
    if (wid == 0) writeOutTraj(s, lane, row0, st + 1, out, fWo, bobsr);
    __syncthreads();
  }
}

extern "C" void kernel_launch(void* const* d_in, const int* in_sizes, int n_in,
                              void* d_out, int out_size, void* d_ws, size_t ws_size,
                              hipStream_t stream) {
  (void)in_sizes; (void)n_in; (void)out_size; (void)d_ws; (void)ws_size;
  const float* latent = (const float*)d_in[0];
  const float* tArr   = (const float*)d_in[1];
  const float* W_l2d  = (const float*)d_in[2];
  const float* b_l2d  = (const float*)d_in[3];
  const float* W1     = (const float*)d_in[4];
  const float* b1     = (const float*)d_in[5];
  const float* W2     = (const float*)d_in[6];
  const float* b2     = (const float*)d_in[7];
  const float* W3     = (const float*)d_in[8];
  const float* b3     = (const float*)d_in[9];
  const float* W_obs  = (const float*)d_in[10];
  const float* b_obs  = (const float*)d_in[11];
  float* out = (float*)d_out;

  ode_rollout_kernel<<<dim3(BATCH / 16), dim3(64), 0, stream>>>(
      latent, tArr, W_l2d, b_l2d, W1, b1, W2, b2, W3, b3, W_obs, b_obs, out);
}
